// PINNNetwork_54820962566939
// MI455X (gfx1250) — matrix operand order repaired and checked by reading, not compiled, not run on hardware
//
#include <hip/hip_runtime.h>

// CDNA5 / gfx1250 WMMA-based fused MLP (Fourier-feature PINN forward pass).
// 32 rows (two 16-row M-tiles) per wave per iteration: each LDS B-fragment
// feeds two independent v_wmma_f32_16x16x32_f16 chains. Activations on the
// TRANS pipe (v_tanh_f32 / v_sin / v_cos) co-execute with XDL WMMA.

typedef __attribute__((ext_vector_type(16))) _Float16 v16h;
typedef __attribute__((ext_vector_type(8)))  _Float16 v8h;
typedef __attribute__((ext_vector_type(8)))  float    v8f;

#define WAVES_PER_BLOCK 4
#define BLOCK_THREADS   (WAVES_PER_BLOCK * 32)
#define GRID_BLOCKS     2048
#define SSTRIDE         72   // 64 cols + 8 f16 pad -> conflict-free lane->bank map

__device__ __forceinline__ float act_tanh(float x) {
#if __has_builtin(__builtin_amdgcn_tanhf)
  return __builtin_amdgcn_tanhf(x);          // v_tanh_f32: 1 TRANS op
#else
  float e = __builtin_amdgcn_exp2f(x * 2.8853900817779268f);
  return (e - 1.0f) * __builtin_amdgcn_rcpf(e + 1.0f);
#endif
}

__device__ __forceinline__ v8f wmma16(v16h a, v16h b, v8f c) {
  return __builtin_amdgcn_wmma_f32_16x16x32_f16(false, a, false, b, (short)0, c,
                                                false, false);
}

// B-fragment (K=32 x N=16, f16): lane n&15 = column N, K = 8*(lane>>4)+e (e<8), 16+8*(lane>>4)+(e-8) (e>=8).
__device__ __forceinline__ v16h load_bfrag(const _Float16* W, int inDim, int nt,
                                           int kt, int lane) {
  const _Float16* p = W + (nt * 16 + (lane & 15)) * inDim + kt * 32 + ((lane >> 4) << 3);
  union { v16h v; v8h h[2]; } u;
  u.h[0] = *(const v8h*)(p);
  u.h[1] = *(const v8h*)(p + 16);
  return u.v;
}

// A-fragment (M=16 x K=32) from row-major staging: lanes 0-15 row M=lane,
// K={0..7,16..23}; lanes 16-31 row M=lane-16, K={8..15,24..31}.
__device__ __forceinline__ v16h load_afrag(const _Float16* S, int kt, int lane) {
  const _Float16* p = S + (lane & 15) * SSTRIDE + kt * 32 + ((lane >> 4) << 3);
  union { v16h v; v8h h[2]; } u;
  u.h[0] = *(const v8h*)(p);
  u.h[1] = *(const v8h*)(p + 16);
  return u.v;
}

// t[e] = x . B_j for the 16 projection indices this lane covers:
// j = jbase + (e&7) + 16*(e>>3) + 8*(lane>>4)
__device__ __forceinline__ void compute_t(float x0, float x1, const float* Bm,
                                          int jbase, int lane, float* t) {
  const int g8 = (lane >> 4) << 3;
#pragma unroll
  for (int e = 0; e < 16; ++e) {
    int j = jbase + (e & 7) + ((e >> 3) << 4) + g8;
    t[e] = x0 * Bm[2 * j] + x1 * Bm[2 * j + 1];
  }
}

// Hardware v_cos/v_sin compute cos/sin(2*pi*x) -> matches reference's 2*pi scale.
__device__ __forceinline__ v16h cos_frag(const float* t) {
  v16h o;
#pragma unroll
  for (int e = 0; e < 16; ++e) o[e] = (_Float16)__builtin_amdgcn_cosf(t[e]);
  return o;
}
__device__ __forceinline__ v16h sin_frag(const float* t) {
  v16h o;
#pragma unroll
  for (int e = 0; e < 16; ++e) o[e] = (_Float16)__builtin_amdgcn_sinf(t[e]);
  return o;
}

__global__ __launch_bounds__(BLOCK_THREADS)
void pinn_fused_kernel(const float* __restrict__ x,   // [N,2]
                       const float* __restrict__ Bm,  // [64,2]
                       const float* __restrict__ W1,  // [64,128]
                       const float* __restrict__ W2,  // [64,64]
                       const float* __restrict__ W3,  // [64,64]
                       const float* __restrict__ W4,  // [64,64]
                       const float* __restrict__ W5,  // [1,64]
                       const float* __restrict__ b1,
                       const float* __restrict__ b2,
                       const float* __restrict__ b3,
                       const float* __restrict__ b4,
                       const float* __restrict__ b5,
                       float* __restrict__ outp,      // [N,1]
                       int nRows) {
  __shared__ _Float16 sW1[64 * 128];
  __shared__ _Float16 sW2[64 * 64];
  __shared__ _Float16 sW3[64 * 64];
  __shared__ _Float16 sW4[64 * 64];
  __shared__ float    sB[64 * 2];
  __shared__ float    sb[4][64];
  __shared__ float    sW5[64];
  __shared__ float    sb5;
  __shared__ _Float16 sStage[WAVES_PER_BLOCK][32 * SSTRIDE];

  // ---- cooperative weight preload: f32 global -> f16 LDS ----
  for (int i = threadIdx.x; i < 64 * 128; i += BLOCK_THREADS) sW1[i] = (_Float16)W1[i];
  for (int i = threadIdx.x; i < 64 * 64; i += BLOCK_THREADS) {
    sW2[i] = (_Float16)W2[i];
    sW3[i] = (_Float16)W3[i];
    sW4[i] = (_Float16)W4[i];
  }
  for (int i = threadIdx.x; i < 64; i += BLOCK_THREADS) {
    sB[2 * i]     = Bm[2 * i];
    sB[2 * i + 1] = Bm[2 * i + 1];
    sb[0][i] = b1[i];
    sb[1][i] = b2[i];
    sb[2][i] = b3[i];
    sb[3][i] = b4[i];
    sW5[i]   = W5[i];
  }
  if (threadIdx.x == 0) sb5 = b5[0];
  __syncthreads();

  const int lane = threadIdx.x & 31;
  const int wave = threadIdx.x >> 5;
  _Float16* S = &sStage[wave][0];

  const long numTiles   = (long)nRows >> 5;  // 32 rows per tile (N % 32 == 0)
  const long tileStride = (long)gridDim.x * WAVES_PER_BLOCK;
  long tile = (long)blockIdx.x * WAVES_PER_BLOCK + wave;

  for (; tile < numTiles; tile += tileStride) {
    const long base = tile << 5;
    const int  n15  = lane & 15;
    const int  mh   = (lane >> 4) << 3;

    // ---- Fourier features, built directly in A-fragment layout ----
    v16h a[2][4];
#pragma unroll
    for (int mt = 0; mt < 2; ++mt) {
      const long row = base + mt * 16 + n15;
      const float x0 = x[row * 2 + 0];
      const float x1 = x[row * 2 + 1];
      float ta[16], tb[16];
      compute_t(x0, x1, sB, 0,  lane, ta);   // shared by frags 0 (cos) and 2 (sin)
      compute_t(x0, x1, sB, 32, lane, tb);   // shared by frags 1 (cos) and 3 (sin)
      a[mt][0] = cos_frag(ta);
      a[mt][1] = cos_frag(tb);
      a[mt][2] = sin_frag(ta);
      a[mt][3] = sin_frag(tb);
    }

    // ---- layer 1: Phi[32x128] @ W1^T[128x64] ----
#pragma unroll
    for (int nt = 0; nt < 4; ++nt) {
      v8f acc[2] = {{}, {}};
#pragma unroll
      for (int kt = 0; kt < 4; ++kt) {
        v16h b = load_bfrag(sW1, 128, nt, kt, lane);   // loaded once, used twice
        acc[0] = wmma16(a[0][kt], b, acc[0]);
        acc[1] = wmma16(a[1][kt], b, acc[1]);
      }
      const float bias = sb[0][nt * 16 + n15];
#pragma unroll
      for (int mt = 0; mt < 2; ++mt) {
        _Float16* Smt = S + mt * 16 * SSTRIDE;
#pragma unroll
        for (int r = 0; r < 8; ++r) {
          float hv = act_tanh(acc[mt][r] + bias);
          Smt[(r + mh) * SSTRIDE + nt * 16 + n15] = (_Float16)hv;
        }
      }
    }

    // ---- layers 2..4: H[32x64] @ W^T[64x64], tanh ----
    const _Float16* Ws[3] = {sW2, sW3, sW4};
    float hreg[2][4][8];  // layer-4 activations kept live for the head
#pragma unroll
    for (int L = 0; L < 3; ++L) {
      v16h ka[2], kb[2];
#pragma unroll
      for (int mt = 0; mt < 2; ++mt) {
        ka[mt] = load_afrag(S + mt * 16 * SSTRIDE, 0, lane);
        kb[mt] = load_afrag(S + mt * 16 * SSTRIDE, 1, lane);
      }
#pragma unroll
      for (int nt = 0; nt < 4; ++nt) {
        v16h b0 = load_bfrag(Ws[L], 64, nt, 0, lane);
        v16h b1 = load_bfrag(Ws[L], 64, nt, 1, lane);
        const float bias = sb[L + 1][nt * 16 + n15];
#pragma unroll
        for (int mt = 0; mt < 2; ++mt) {
          v8f acc = {};
          acc = wmma16(ka[mt], b0, acc);
          acc = wmma16(kb[mt], b1, acc);
          _Float16* Smt = S + mt * 16 * SSTRIDE;
#pragma unroll
          for (int r = 0; r < 8; ++r) {
            float hv = act_tanh(acc[r] + bias);
            hreg[mt][nt][r] = hv;
            if (L < 2) Smt[(r + mh) * SSTRIDE + nt * 16 + n15] = (_Float16)hv;
          }
        }
      }
    }

    // ---- head: out[m] = b5 + sum_f w5[f] * h4[m][f] ----
    const float bb = sb5;
#pragma unroll
    for (int mt = 0; mt < 2; ++mt) {
      float partial[8];
#pragma unroll
      for (int r = 0; r < 8; ++r) partial[r] = 0.0f;
#pragma unroll
      for (int nt = 0; nt < 4; ++nt) {
        const float w = sW5[nt * 16 + n15];
#pragma unroll
        for (int r = 0; r < 8; ++r) partial[r] += hreg[mt][nt][r] * w;
      }
      // reduce across the 16 lanes of each half-wave (halves hold different rows)
#pragma unroll
      for (int mask = 1; mask < 16; mask <<= 1) {
#pragma unroll
        for (int r = 0; r < 8; ++r)
          partial[r] += __shfl_xor(partial[r], mask, 32);
      }
#pragma unroll
      for (int r = 0; r < 8; ++r) {
        if (n15 == r) outp[base + mt * 16 + r + mh] = partial[r] + bb;
      }
    }
  }
}

extern "C" void kernel_launch(void* const* d_in, const int* in_sizes, int n_in,
                              void* d_out, int out_size, void* d_ws, size_t ws_size,
                              hipStream_t stream) {
  const float* x  = (const float*)d_in[0];   // [N,2]
  const float* Bm = (const float*)d_in[1];   // [64,2]
  const float* W1 = (const float*)d_in[2];   // [64,128]
  const float* W2 = (const float*)d_in[3];   // [64,64]
  const float* W3 = (const float*)d_in[4];   // [64,64]
  const float* W4 = (const float*)d_in[5];   // [64,64]
  const float* W5 = (const float*)d_in[6];   // [1,64]
  const float* b1 = (const float*)d_in[7];
  const float* b2 = (const float*)d_in[8];
  const float* b3 = (const float*)d_in[9];
  const float* b4 = (const float*)d_in[10];
  const float* b5 = (const float*)d_in[11];
  float* outp = (float*)d_out;
  const int nRows = in_sizes[0] / 2;

  pinn_fused_kernel<<<GRID_BLOCKS, BLOCK_THREADS, 0, stream>>>(
      x, Bm, W1, W2, W3, W4, W5, b1, b2, b3, b4, b5, outp, nRows);
}
